// RESK1_40956808135034
// MI455X (gfx1250) — matrix operand order repaired and checked by reading, not compiled, not run on hardware
//
#include <hip/hip_runtime.h>

typedef __attribute__((ext_vector_type(16))) __bf16 v16bf;
typedef __attribute__((ext_vector_type(8)))  float  v8f;

union V16BF {
    v16bf v;
    __bf16 e[16];
    uint4 u[2];
};

// LDS row pitch for the bf16 weight panel: 128 data + 8 pad ushorts.
// 272B = 68 dwords -> lane-to-lane bank offset of 4 banks; the 16 lanes of a
// half-wave cover disjoint bank quartets for ds_load_b128 (no conflicts),
// and row starts stay 16B-aligned for uint4 access.
#define BPITCH 136

// ---------------------------------------------------------------------------
// Transpose fp32 W[K][outC] -> bf16 Wt[outCpad][K], zero-padding extra cols.
// ---------------------------------------------------------------------------
__global__ void wtrans_kernel(const float* __restrict__ W,
                              unsigned short* __restrict__ Wt,
                              int K, int outC, int outCpad) {
    int i = blockIdx.x * blockDim.x + threadIdx.x;
    if (i >= outCpad * K) return;
    int o = i / K;
    int k = i - o * K;
    float v = (o < outC) ? W[(size_t)k * outC + o] : 0.0f;
    __bf16 bv = (__bf16)v;  // native cvt, RNE
    Wt[(size_t)o * K + k] = __builtin_bit_cast(unsigned short, bv);
}

// ---------------------------------------------------------------------------
// WMMA GEMM:  Cout[M][ldc] = A[M][128] * Wt^T   (Wt is bf16 [NCOLT*16][128])
// 256 threads = 8 waves. colgroups = NCOLT/CT_PER_WAVE; ROWT*colgroups == 8.
// Each wave owns 16 rows x (CT_PER_WAVE*16) cols; K=128 in 4 steps of 32, so
// each converted A fragment feeds CT_PER_WAVE back-to-back WMMAs.
// ---------------------------------------------------------------------------
template <int NCOLT, int CT_PER_WAVE, int ROWT_PER_BLOCK>
__global__ __launch_bounds__(256) void gemm_wmma_kernel(
    const float* __restrict__ A,
    const unsigned short* __restrict__ Wt,
    float* __restrict__ Cout,
    int M, int ldc, int ncols) {
    static_assert((NCOLT / CT_PER_WAVE) * ROWT_PER_BLOCK == 8, "wave mapping");
    __shared__ __align__(16) unsigned short Blds[NCOLT * 16 * BPITCH];

    const int tid = threadIdx.x;

    // Cooperative stage of the weight panel into padded LDS rows.
    {
        const int n4 = NCOLT * 16 * 16;  // uint4 count (16 per 128-ushort row)
        for (int i = tid; i < n4; i += 256) {
            int r = i >> 4;
            int j = i & 15;
            *(uint4*)(Blds + r * BPITCH + j * 8) =
                *(const uint4*)(Wt + r * 128 + j * 8);
        }
    }
    __syncthreads();

    const int wave = tid >> 5;
    const int lane = tid & 31;
    const int rowt = wave % ROWT_PER_BLOCK;
    const int colg = wave / ROWT_PER_BLOCK;
    const int lm = lane & 15;   // M (for A) / N (for B,C)
    const int lh = lane >> 4;   // K-half selector

    const int rowbase = blockIdx.x * (ROWT_PER_BLOCK * 16) + rowt * 16;

    int arow = rowbase + lm;
    if (arow >= M) arow = M - 1;            // clamp; stores are guarded
    const float* __restrict__ arowp = A + (size_t)arow * 128;

    v8f zero = {};
    v8f acc[CT_PER_WAVE];
#pragma unroll
    for (int c = 0; c < CT_PER_WAVE; ++c) acc[c] = zero;

#pragma unroll
    for (int ks = 0; ks < 4; ++ks) {
        // ---- A fragment: elems 0..15 = K kb..kb+15
        const int kb = ks * 32 + lh * 16;
        float4 x0 = *(const float4*)(arowp + kb);
        float4 x1 = *(const float4*)(arowp + kb + 4);
        float4 x2 = *(const float4*)(arowp + kb + 8);
        float4 x3 = *(const float4*)(arowp + kb + 12);
        V16BF af;
        af.e[0] = (__bf16)x0.x;  af.e[1] = (__bf16)x0.y;
        af.e[2] = (__bf16)x0.z;  af.e[3] = (__bf16)x0.w;
        af.e[4] = (__bf16)x1.x;  af.e[5] = (__bf16)x1.y;
        af.e[6] = (__bf16)x1.z;  af.e[7] = (__bf16)x1.w;
        af.e[8] = (__bf16)x2.x;  af.e[9] = (__bf16)x2.y;
        af.e[10] = (__bf16)x2.z; af.e[11] = (__bf16)x2.w;
        af.e[12] = (__bf16)x3.x; af.e[13] = (__bf16)x3.y;
        af.e[14] = (__bf16)x3.z; af.e[15] = (__bf16)x3.w;

#pragma unroll
        for (int ct = 0; ct < CT_PER_WAVE; ++ct) {
            // ---- B fragment from LDS: column n = tile*16+lm, 16 consecutive K
            const int bcol = (colg * CT_PER_WAVE + ct) * 16 + lm;
            const int bk = ks * 32 + lh * 16;
            const uint4* bp = (const uint4*)(Blds + bcol * BPITCH + bk);
            V16BF bf;
            bf.u[0] = bp[0];
            bf.u[1] = bp[1];
            acc[ct] = __builtin_amdgcn_wmma_f32_16x16x32_bf16(
                false, af.v, false, bf.v, (short)0, acc[ct], false, false);
        }
    }

    // ---- store: VGPR i holds M = i + 8*lh, N = lm
#pragma unroll
    for (int ct = 0; ct < CT_PER_WAVE; ++ct) {
        const int col = (colg * CT_PER_WAVE + ct) * 16 + lm;
        if (col < ncols) {
#pragma unroll
            for (int i = 0; i < 8; ++i) {
                const int row = rowbase + i + 8 * lh;
                if (row < M) Cout[(size_t)row * ldc + col] = acc[ct][i];
            }
        }
    }
}

// ---------------------------------------------------------------------------
// Edge-parallel weighted scatter-add: agg[tgt[e]] += h[src[e]] * w[e]
// One thread per (edge, 4-column group). Random traffic stays in L2
// (h and agg tables are 51.2 MB each < 192 MB L2).
// ---------------------------------------------------------------------------
template <int C>
__global__ void agg_edges_kernel(const float* __restrict__ h,
                                 const int* __restrict__ srcI,
                                 const int* __restrict__ tgtI,
                                 const float* __restrict__ wE,
                                 float* __restrict__ agg, int E) {
    constexpr int G = C / 4;
    unsigned int idx = blockIdx.x * blockDim.x + threadIdx.x;
    if (idx >= (unsigned int)E * G) return;
    unsigned int e = idx / G;
    unsigned int g = (idx - e * G) * 4;
    int s = srcI[e];
    int t = tgtI[e];
    float we = wE[e];
    float4 hv = *(const float4*)(h + (size_t)s * C + g);
    float* dst = agg + (size_t)t * C + g;
    atomicAdd(dst + 0, hv.x * we);
    atomicAdd(dst + 1, hv.y * we);
    atomicAdd(dst + 2, hv.z * we);
    atomicAdd(dst + 3, hv.w * we);
}

// ---------------------------------------------------------------------------
// Epilogues
// ---------------------------------------------------------------------------
__global__ void bias_relu_kernel(const float* __restrict__ agg,
                                 const float* __restrict__ b,
                                 float* __restrict__ out, int total, int C) {
    int i = blockIdx.x * blockDim.x + threadIdx.x;
    if (i >= total) return;
    int c = i & (C - 1);
    float v = agg[i] + b[c];
    out[i] = v > 0.0f ? v : 0.0f;
}

__global__ void bias_relu_res_kernel(const float* __restrict__ agg,
                                     const float* __restrict__ b,
                                     float* __restrict__ act, int total, int C) {
    int i = blockIdx.x * blockDim.x + threadIdx.x;
    if (i >= total) return;
    int c = i & (C - 1);
    float v = agg[i] + b[c];
    v = v > 0.0f ? v : 0.0f;
    act[i] = v + act[i];
}

__global__ void logsoftmax40_kernel(const float* __restrict__ agg,
                                    const float* __restrict__ b,
                                    float* __restrict__ out, int Nn) {
    int n = blockIdx.x * blockDim.x + threadIdx.x;
    if (n >= Nn) return;
    const float* p = agg + (size_t)n * 40;
    float v[40];
    float mx = -3.402823e38f;
#pragma unroll
    for (int c = 0; c < 40; ++c) {
        v[c] = p[c] + b[c];
        mx = fmaxf(mx, v[c]);
    }
    float s = 0.0f;
#pragma unroll
    for (int c = 0; c < 40; ++c) s += __expf(v[c] - mx);
    float l = __logf(s) + mx;
    float* o = out + (size_t)n * 40;
#pragma unroll
    for (int c = 0; c < 40; ++c) o[c] = v[c] - l;
}

// ---------------------------------------------------------------------------
extern "C" void kernel_launch(void* const* d_in, const int* in_sizes, int n_in,
                              void* d_out, int out_size, void* d_ws, size_t ws_size,
                              hipStream_t stream) {
    const float* x  = (const float*)d_in[0];
    const int* srcI = (const int*)d_in[1];
    const int* tgtI = (const int*)d_in[2];
    const float* wE = (const float*)d_in[3];
    const float* W0 = (const float*)d_in[4];
    const float* b0 = (const float*)d_in[5];
    const float* W1 = (const float*)d_in[6];
    const float* b1 = (const float*)d_in[7];
    const float* W2 = (const float*)d_in[8];
    const float* b2 = (const float*)d_in[9];
    const float* W3 = (const float*)d_in[10];
    const float* b3 = (const float*)d_in[11];
    float* out = (float*)d_out;

    const int Nn = 100000, Ee = 1600000;

    float* act = (float*)d_ws;                  // [N,128]
    float* h   = act + (size_t)Nn * 128;        // [N,128] (reused as [N,40] for L3)
    float* agg = h + (size_t)Nn * 128;          // [N,128] (reused as [N,40] for L3)
    unsigned short* Wt0 = (unsigned short*)(agg + (size_t)Nn * 128);
    unsigned short* Wt1 = Wt0 + 128 * 128;
    unsigned short* Wt2 = Wt1 + 128 * 128;
    unsigned short* Wt3 = Wt2 + 128 * 128;      // 48x128 used

    // --- weight prep (tiny) ---
    wtrans_kernel<<<(128 * 128 + 255) / 256, 256, 0, stream>>>(W0, Wt0, 128, 128, 128);
    wtrans_kernel<<<(128 * 128 + 255) / 256, 256, 0, stream>>>(W1, Wt1, 128, 128, 128);
    wtrans_kernel<<<(128 * 128 + 255) / 256, 256, 0, stream>>>(W2, Wt2, 128, 128, 128);
    wtrans_kernel<<<(128 * 48 + 255) / 256, 256, 0, stream>>>(W3, Wt3, 128, 40, 48);

    const int gemmBlocks128 = (Nn + 127) / 128;
    const int gemmBlocks48  = (Nn + 127) / 128;
    const int aggBlocks128  = (Ee * 32 + 255) / 256;
    const int aggBlocks40   = (Ee * 10 + 255) / 256;
    const int ewBlocks      = (Nn * 128 + 255) / 256;

    // --- layer 0 ---
    gemm_wmma_kernel<8, 8, 8><<<gemmBlocks128, 256, 0, stream>>>(x, Wt0, h, Nn, 128, 128);
    hipMemsetAsync(agg, 0, (size_t)Nn * 128 * sizeof(float), stream);
    agg_edges_kernel<128><<<aggBlocks128, 256, 0, stream>>>(h, srcI, tgtI, wE, agg, Ee);
    bias_relu_kernel<<<ewBlocks, 256, 0, stream>>>(agg, b0, act, Nn * 128, 128);

    // --- layer 1 (residual) ---
    gemm_wmma_kernel<8, 8, 8><<<gemmBlocks128, 256, 0, stream>>>(act, Wt1, h, Nn, 128, 128);
    hipMemsetAsync(agg, 0, (size_t)Nn * 128 * sizeof(float), stream);
    agg_edges_kernel<128><<<aggBlocks128, 256, 0, stream>>>(h, srcI, tgtI, wE, agg, Ee);
    bias_relu_res_kernel<<<ewBlocks, 256, 0, stream>>>(agg, b1, act, Nn * 128, 128);

    // --- layer 2 (residual) ---
    gemm_wmma_kernel<8, 8, 8><<<gemmBlocks128, 256, 0, stream>>>(act, Wt2, h, Nn, 128, 128);
    hipMemsetAsync(agg, 0, (size_t)Nn * 128 * sizeof(float), stream);
    agg_edges_kernel<128><<<aggBlocks128, 256, 0, stream>>>(h, srcI, tgtI, wE, agg, Ee);
    bias_relu_res_kernel<<<ewBlocks, 256, 0, stream>>>(agg, b2, act, Nn * 128, 128);

    // --- layer 3: 128 -> 40, then log-softmax ---
    gemm_wmma_kernel<3, 3, 8><<<gemmBlocks48, 256, 0, stream>>>(act, Wt3, h, Nn, 40, 40);
    hipMemsetAsync(agg, 0, (size_t)Nn * 40 * sizeof(float), stream);
    agg_edges_kernel<40><<<aggBlocks40, 256, 0, stream>>>(h, srcI, tgtI, wE, agg, Ee);
    logsoftmax40_kernel<<<(Nn + 255) / 256, 256, 0, stream>>>(agg, b3, out, Nn);
}
